// MultiheadSelfAttention_29643864277514
// MI455X (gfx1250) — matrix operand order repaired and checked by reading, not compiled, not run on hardware
//
#include <hip/hip_runtime.h>
#include <cmath>

// ---------------------------------------------------------------------------
// Multi-head causal self-attention, MI455X (gfx1250).
// bf16 WMMA for all 4 GEMMs + both attention matmuls; f32 accumulate.
// Async-to-LDS double-buffered weight staging in the GEMMs (ASYNCcnt path).
// Softmax row-sums computed on the matrix unit (P x ones WMMA).
// B=2, T=2048, D=1024, H=16, dh=64.
// ---------------------------------------------------------------------------

#define B_SZ   2
#define T_LEN  2048
#define D_MOD  1024
#define N_HEAD 16
#define DH     64
#define M_ALL  (B_SZ * T_LEN)   // 4096 rows for the big GEMMs

typedef __attribute__((ext_vector_type(16))) __bf16 bf16x16;
typedef __attribute__((ext_vector_type(8)))  float  f32x8;

// D = A(16x32) * B(32x16) + C, bf16 in / f32 accumulate
__device__ __forceinline__ f32x8 wmma_bf16(bf16x16 a, bf16x16 b, f32x8 c) {
  return __builtin_amdgcn_wmma_f32_16x16x32_bf16(
      false, a, false, b, (short)0, c, false, false);
}

// A-fragment (16x32, 16-bit): lane holds row M = lane&15.
// elements 0..15 -> K = 16*laneHi + 0..15 (contiguous 32 bytes)
__device__ __forceinline__ bf16x16 load_fragA(const __bf16* p, int laneHi) {
  const __bf16* q = p + laneHi * 16;
  bf16x16 f;
  reinterpret_cast<float4*>(&f)[0] = *reinterpret_cast<const float4*>(q);
  reinterpret_cast<float4*>(&f)[1] = *reinterpret_cast<const float4*>(q + 8);
  return f;
}

// B-fragment (32x16, 16-bit): lane holds col N = lane&15.
// elements 0..15 -> K = 16*laneHi + 0..15 (contiguous 32 bytes)
__device__ __forceinline__ bf16x16 load_fragB(const __bf16* colp, int laneHi) {
  const __bf16* q = colp + laneHi * 16;
  bf16x16 f;
  reinterpret_cast<float4*>(&f)[0] = *reinterpret_cast<const float4*>(q);
  reinterpret_cast<float4*>(&f)[1] = *reinterpret_cast<const float4*>(q + 8);
  return f;
}

__device__ __forceinline__ bf16x16 ones_frag() {
  bf16x16 f;
#pragma unroll
  for (int e = 0; e < 16; ++e) f[e] = (__bf16)1.0f;
  return f;
}

// ---------------------------------------------------------------------------
// gfx1250 async global->LDS copy (16B per lane), ASYNCcnt-tracked.
// Builtin prototype (from hipcc diagnostic): param0 = <4 x i32> AS(1)*,
// param1 = <4 x i32> AS(3)*, then imm offset, cpol. Flat LDS addresses carry
// the LDS byte offset in the low 32 bits (ISA 10.2), so inttoptr reconstructs
// both address-space pointers.
// ---------------------------------------------------------------------------
typedef int v4i __attribute__((vector_size(16)));
typedef __attribute__((address_space(1))) v4i as1_v4i;
typedef __attribute__((address_space(3))) v4i as3_v4i;

__device__ __forceinline__ void async_copy_b128(const __bf16* g, __bf16* lds) {
#if __has_builtin(__builtin_amdgcn_global_load_async_to_lds_b128)
  __builtin_amdgcn_global_load_async_to_lds_b128(
      (as1_v4i*)(uintptr_t)g,
      (as3_v4i*)(unsigned int)(uintptr_t)lds, 0, 0);
#else
  *reinterpret_cast<float4*>(lds) = *reinterpret_cast<const float4*>(g);
#endif
}

__device__ __forceinline__ void wait_async_le2() {
#if __has_builtin(__builtin_amdgcn_s_wait_asynccnt)
  __builtin_amdgcn_s_wait_asynccnt(2);
#else
  asm volatile("s_wait_asynccnt 2" ::: "memory");
#endif
}
__device__ __forceinline__ void wait_async_0() {
#if __has_builtin(__builtin_amdgcn_s_wait_asynccnt)
  __builtin_amdgcn_s_wait_asynccnt(0);
#else
  asm volatile("s_wait_asynccnt 0" ::: "memory");
#endif
}

// ---------------------------------------------------------------------------
// f32 -> bf16 cast
// ---------------------------------------------------------------------------
__global__ void cast_bf16_kernel(const float* __restrict__ x,
                                 __bf16* __restrict__ y, int n) {
  int i = blockIdx.x * blockDim.x + threadIdx.x;
  if (i < n) y[i] = (__bf16)x[i];
}

// ---------------------------------------------------------------------------
// GEMM: C[m,n] = sum_k A[m,k] * W[n,k]  (A row-major [M,K], W row-major [N,K])
// Block = 128 threads (4 waves) computing a 128x64 tile; each wave owns 32x64.
// Per k-step (32): W tile (64x32 = 4KB) async-staged to LDS, double-buffered;
// 8 WMMA per wave against 4 global b128 (A) + 8 ds b128 (B, shared).
// MODE 0: bf16 [M,N] | MODE 1: bf16 V^T per batch [b,N,T] | MODE 2: f32 [M,N]
// ---------------------------------------------------------------------------
template <int MODE>
__global__ __launch_bounds__(128) void gemm_bf16_kernel(
    const __bf16* __restrict__ A, const __bf16* __restrict__ W,
    void* __restrict__ outp, int M, int N, int K) {
  __shared__ __align__(16) __bf16 Bt[2][64 * 32];  // [col][k], 2 x 4KB

  const int tid    = threadIdx.x;
  const int lane   = tid & 31;
  const int wv     = tid >> 5;
  const int laneLo = lane & 15;
  const int laneHi = lane >> 4;
  const int m0 = blockIdx.x * 128 + wv * 32;
  const int n0 = blockIdx.y * 64;
  const int nK = K / 32;

  f32x8 c[2][4] = {};
  const __bf16* arow0 = A + (size_t)(m0 + laneLo) * K;
  const __bf16* arow1 = arow0 + (size_t)16 * K;

  // stage W[(n0..n0+63), kstep..kstep+31] -> Bt[buf]; 2 x 16B per thread
  auto issue = [&](int buf, int kstep) {
#pragma unroll
    for (int j = 0; j < 2; ++j) {
      const int chunk = tid * 2 + j;      // 0..255
      const int col   = chunk >> 2;       // 0..63
      const int ko    = (chunk & 3) * 8;  // 0,8,16,24
      async_copy_b128(W + (size_t)(n0 + col) * K + kstep + ko,
                      &Bt[buf][col * 32 + ko]);
    }
  };

  issue(0, 0);
  for (int ki = 0; ki < nK; ++ki) {
    const int cur = ki & 1;
    if (ki + 1 < nK) {
      issue(cur ^ 1, (ki + 1) * 32);
      wait_async_le2();   // async loads retire in order -> batch ki is in LDS
    } else {
      wait_async_0();
    }
    __syncthreads();      // LDS tile visible to all 4 waves

    const int k = ki * 32;
    const bf16x16 a0 = load_fragA(arow0 + k, laneHi);
    const bf16x16 a1 = load_fragA(arow1 + k, laneHi);
#pragma unroll
    for (int n = 0; n < 4; ++n) {
      const bf16x16 b = load_fragB(&Bt[cur][(n * 16 + laneLo) * 32], laneHi);
      c[0][n] = wmma_bf16(a0, b, c[0][n]);
      c[1][n] = wmma_bf16(a1, b, c[1][n]);
    }
    __syncthreads();      // everyone done reading before buffer is re-issued
  }

#pragma unroll
  for (int part = 0; part < 2; ++part) {
#pragma unroll
    for (int n = 0; n < 4; ++n) {
#pragma unroll
      for (int r = 0; r < 8; ++r) {
        const int row = m0 + part * 16 + r + 8 * laneHi;
        const int col = n0 + n * 16 + laneLo;
        const float v = c[part][n][r];
        if (MODE == 0) {
          ((__bf16*)outp)[(size_t)row * N + col] = (__bf16)v;
        } else if (MODE == 1) {
          const int bb = row / T_LEN, t = row % T_LEN;
          ((__bf16*)outp)[((size_t)bb * N + col) * T_LEN + t] = (__bf16)v;
        } else {
          ((float*)outp)[(size_t)row * N + col] = v;
        }
      }
    }
  }
}

// ---------------------------------------------------------------------------
// RoPE (interleaved pairs), in place on bf16 Q and K.
// The attention scale 1/sqrt(dh) = 0.125 is folded into Q here so the
// flash-attention inner loop needs no per-score multiply.
// ---------------------------------------------------------------------------
__global__ void rope_kernel(__bf16* __restrict__ Q, __bf16* __restrict__ K) {
  const int total = B_SZ * T_LEN * (D_MOD / 2);
  int i = blockIdx.x * blockDim.x + threadIdx.x;
  if (i >= total) return;
  const int pair = i % (D_MOD / 2);
  const int t    = (i / (D_MOD / 2)) % T_LEN;
  const int b    = i / (D_MOD / 2 * T_LEN);
  const int h    = pair / (DH / 2);
  const int k    = pair % (DH / 2);

  const float freq = __expf(-(float)k * (9.210340371976184f / 32.0f));
  float s, c;
  __sincosf((float)t * freq, &s, &c);

  const size_t off = ((size_t)(b * T_LEN + t)) * D_MOD + h * DH + 2 * k;
  const float q0 = (float)Q[off], q1 = (float)Q[off + 1];
  Q[off]     = (__bf16)((c * q0 - s * q1) * 0.125f);
  Q[off + 1] = (__bf16)((s * q0 + c * q1) * 0.125f);
  const float k0 = (float)K[off], k1 = (float)K[off + 1];
  K[off]     = (__bf16)(c * k0 - s * k1);
  K[off + 1] = (__bf16)(s * k0 + c * k1);
}

// ---------------------------------------------------------------------------
// One flash-attention key step (32 keys):
//   4 WMMA (QK^T over dh=64), row-max via shfl_xor, exp, P->LDS relayout,
//   1 WMMA (P x ones -> exact bf16 row sums), 4 WMMA (P x V).
// MASKED instantiation is used only for the single diagonal tile.
// ---------------------------------------------------------------------------
template <bool MASKED>
__device__ __forceinline__ void attn_step(
    int kb, int m0, int b, int hh, int laneLo, int laneHi,
    const __bf16* __restrict__ K, const __bf16* __restrict__ Vt,
    const bf16x16& aQ0, const bf16x16& aQ1, const bf16x16& ones,
    f32x8 (&acc)[4], float (&mrow)[8], float (&lrow)[8],
    __bf16 (*P)[32]) {
  f32x8 S0 = {}, S1 = {};
  {
    const __bf16* kr0 = K + ((size_t)(b * T_LEN + kb + laneLo)) * D_MOD + hh * DH;
    const __bf16* kr1 = kr0 + (size_t)16 * D_MOD;
    S0 = wmma_bf16(aQ0, load_fragB(kr0 + 0,  laneHi), S0);
    S0 = wmma_bf16(aQ1, load_fragB(kr0 + 32, laneHi), S0);
    S1 = wmma_bf16(aQ0, load_fragB(kr1 + 0,  laneHi), S1);
    S1 = wmma_bf16(aQ1, load_fragB(kr1 + 32, laneHi), S1);
  }

  float corr[8];
#pragma unroll
  for (int r = 0; r < 8; ++r) {
    float x0 = S0[r];
    float x1 = S1[r];
    if (MASKED) {
      const int row = m0 + r + 8 * laneHi;
      x0 = (kb + laneLo      <= row) ? x0 : -1e30f;
      x1 = (kb + laneLo + 16 <= row) ? x1 : -1e30f;
    }
    float mx = fmaxf(x0, x1);
    mx = fmaxf(mx, __shfl_xor(mx, 1));
    mx = fmaxf(mx, __shfl_xor(mx, 2));
    mx = fmaxf(mx, __shfl_xor(mx, 4));
    mx = fmaxf(mx, __shfl_xor(mx, 8));
    const float mnew = fmaxf(mrow[r], mx);
    corr[r] = __expf(mrow[r] - mnew);
    mrow[r] = mnew;
    const float p0 = __expf(x0 - mnew);
    const float p1 = __expf(x1 - mnew);
    acc[0][r] *= corr[r]; acc[1][r] *= corr[r];
    acc[2][r] *= corr[r]; acc[3][r] *= corr[r];
    P[r + 8 * laneHi][laneLo]      = (__bf16)p0;
    P[r + 8 * laneHi][laneLo + 16] = (__bf16)p1;
  }

  // same-wave LDS RAW: compiler inserts s_wait_dscnt; no barrier needed
  const bf16x16 aP = load_fragA(&P[laneLo][0], laneHi);

  // Row sums on the matrix unit: every column of (P x ones) is sum_k P[q,k],
  // exactly consistent with the bf16 P used for P*V below.
  f32x8 rsum = {};
  rsum = wmma_bf16(aP, ones, rsum);
#pragma unroll
  for (int r = 0; r < 8; ++r) lrow[r] = lrow[r] * corr[r] + rsum[r];

#pragma unroll
  for (int n = 0; n < 4; ++n) {
    const __bf16* vr =
        Vt + ((size_t)b * D_MOD + hh * DH + n * 16 + laneLo) * T_LEN + kb;
    acc[n] = wmma_bf16(aP, load_fragB(vr, laneHi), acc[n]);
  }
}

// ---------------------------------------------------------------------------
// Causal flash attention. One wave = one 16-query tile of one (b, h).
// Q (pre-scaled), K: bf16 [B,T,D]; Vt: bf16 [B,D,T]; O: bf16 [B,T,D].
// ---------------------------------------------------------------------------
__global__ __launch_bounds__(128) void attn_kernel(
    const __bf16* __restrict__ Q, const __bf16* __restrict__ K,
    const __bf16* __restrict__ Vt, __bf16* __restrict__ O) {
  __shared__ __align__(16) __bf16 Plds[4][16][32];  // per-wave P staging

  const int lane   = threadIdx.x & 31;
  const int wv     = threadIdx.x >> 5;
  const int laneLo = lane & 15;
  const int laneHi = lane >> 4;
  const int m0 = (blockIdx.x * 4 + wv) * 16;  // query tile base
  const int hh = blockIdx.y;
  const int b  = blockIdx.z;

  const __bf16* qrow = Q + ((size_t)(b * T_LEN + m0 + laneLo)) * D_MOD + hh * DH;
  const bf16x16 aQ0 = load_fragA(qrow + 0, laneHi);
  const bf16x16 aQ1 = load_fragA(qrow + 32, laneHi);
  const bf16x16 ones = ones_frag();

  f32x8 acc[4] = {};
  float mrow[8], lrow[8];
#pragma unroll
  for (int r = 0; r < 8; ++r) { mrow[r] = -1e30f; lrow[r] = 0.0f; }

  // Full (never-masked) tiles: kb+31 <= m0  <=>  kb < (m0 & ~31)
  const int kbFastEnd = m0 & ~31;
  for (int kb = 0; kb < kbFastEnd; kb += 32) {
    // prefetch next K/V key tile into near caches (global_prefetch_b8)
    __builtin_prefetch(
        K + ((size_t)(b * T_LEN + kb + 32 + lane)) * D_MOD + hh * DH, 0, 3);
    __builtin_prefetch(
        Vt + ((size_t)b * D_MOD + hh * DH + lane) * T_LEN + kb + 32, 0, 3);
    attn_step<false>(kb, m0, b, hh, laneLo, laneHi, K, Vt,
                     aQ0, aQ1, ones, acc, mrow, lrow, Plds[wv]);
  }
  // Exactly one diagonal (masked) tile at kb = m0 & ~31
  attn_step<true>(kbFastEnd, m0, b, hh, laneLo, laneHi, K, Vt,
                  aQ0, aQ1, ones, acc, mrow, lrow, Plds[wv]);

#pragma unroll
  for (int r = 0; r < 8; ++r) {
    const int row = m0 + r + 8 * laneHi;
    const float inv = 1.0f / lrow[r];
    __bf16* orow = O + ((size_t)(b * T_LEN + row)) * D_MOD + hh * DH + laneLo;
    orow[0]  = (__bf16)(acc[0][r] * inv);
    orow[16] = (__bf16)(acc[1][r] * inv);
    orow[32] = (__bf16)(acc[2][r] * inv);
    orow[48] = (__bf16)(acc[3][r] * inv);
  }
}

// ---------------------------------------------------------------------------
// Host launcher
// ---------------------------------------------------------------------------
extern "C" void kernel_launch(void* const* d_in, const int* in_sizes, int n_in,
                              void* d_out, int out_size, void* d_ws, size_t ws_size,
                              hipStream_t stream) {
  (void)in_sizes; (void)n_in; (void)out_size; (void)ws_size;
  const float* h_f  = (const float*)d_in[0];
  const float* wq_f = (const float*)d_in[1];
  const float* wk_f = (const float*)d_in[2];
  const float* wv_f = (const float*)d_in[3];
  const float* wo_f = (const float*)d_in[4];
  float* out = (float*)d_out;

  char* ws = (char*)d_ws;                              // 48 MB total
  __bf16* hBF  = (__bf16*)(ws + (size_t)0);            // 8 MB
  __bf16* wqBF = (__bf16*)(ws + (size_t)8  * 1048576); // 2 MB each
  __bf16* wkBF = wqBF + 1048576;
  __bf16* wvBF = wkBF + 1048576;
  __bf16* woBF = wvBF + 1048576;
  __bf16* Qbf  = (__bf16*)(ws + (size_t)16 * 1048576); // [B,T,D]
  __bf16* Kbf  = (__bf16*)(ws + (size_t)24 * 1048576); // [B,T,D]
  __bf16* Vt   = (__bf16*)(ws + (size_t)32 * 1048576); // [B,D,T]
  __bf16* Abf  = (__bf16*)(ws + (size_t)40 * 1048576); // attn out [B,T,D]

  const int nH = M_ALL * D_MOD;
  const int nW = D_MOD * D_MOD;

  cast_bf16_kernel<<<(nH + 255) / 256, 256, 0, stream>>>(h_f,  hBF,  nH);
  cast_bf16_kernel<<<(nW + 255) / 256, 256, 0, stream>>>(wq_f, wqBF, nW);
  cast_bf16_kernel<<<(nW + 255) / 256, 256, 0, stream>>>(wk_f, wkBF, nW);
  cast_bf16_kernel<<<(nW + 255) / 256, 256, 0, stream>>>(wv_f, wvBF, nW);
  cast_bf16_kernel<<<(nW + 255) / 256, 256, 0, stream>>>(wo_f, woBF, nW);

  dim3 ggrid(M_ALL / 128, D_MOD / 64);
  gemm_bf16_kernel<0><<<ggrid, 128, 0, stream>>>(hBF, wqBF, Qbf, M_ALL, D_MOD, D_MOD);
  gemm_bf16_kernel<0><<<ggrid, 128, 0, stream>>>(hBF, wkBF, Kbf, M_ALL, D_MOD, D_MOD);
  gemm_bf16_kernel<1><<<ggrid, 128, 0, stream>>>(hBF, wvBF, Vt,  M_ALL, D_MOD, D_MOD);

  const int nR = B_SZ * T_LEN * (D_MOD / 2);
  rope_kernel<<<(nR + 255) / 256, 256, 0, stream>>>(Qbf, Kbf);

  attn_kernel<<<dim3(T_LEN / 64, N_HEAD, B_SZ), 128, 0, stream>>>(Qbf, Kbf, Vt, Abf);

  gemm_bf16_kernel<2><<<ggrid, 128, 0, stream>>>(Abf, woBF, out, M_ALL, D_MOD, D_MOD);
}
